// MoeFcTokensParallel_31275951850268
// MI455X (gfx1250) — matrix operand order repaired and checked by reading, not compiled, not run on hardware
//
#include <hip/hip_runtime.h>
#include <hip/hip_bf16.h>

// ---------------- problem constants ----------------
#define PB 8      // batch
#define PS 2048   // seq
#define PD 1024   // d_model
#define PO 1024   // d_out per layer
#define PE 16     // experts
#define PK 256    // top-k tokens per (b,e)
#define PBE (PB * PE)

typedef __attribute__((ext_vector_type(16))) __bf16 v16bf;
typedef __attribute__((ext_vector_type(8)))  float  v8f;

// fp32 -> bf16, round-to-nearest-even
static __device__ __forceinline__ unsigned short f2bf(float f) {
  unsigned u = __float_as_uint(f);
  unsigned r = u + 0x7FFFu + ((u >> 16) & 1u);
  return (unsigned short)(r >> 16);
}

union FragU {
  v16bf v;
  uint4 q[2];
};

// two 16B LDS chunks -> one 16-element bf16 fragment (ds_load_b128 x2)
static __device__ __forceinline__ v16bf load_frag(const unsigned short* p0,
                                                  const unsigned short* p1) {
  FragU u;
  u.q[0] = *(const uint4*)p0;
  u.q[1] = *(const uint4*)p1;
  return u.v;
}

// ---------------- 1. zero the output ----------------
__global__ void zero_kernel(float4* __restrict__ p, int n4) {
  int i = blockIdx.x * blockDim.x + threadIdx.x;
  int stride = gridDim.x * blockDim.x;
  float4 z; z.x = 0.f; z.y = 0.f; z.z = 0.f; z.w = 0.f;
  for (; i < n4; i += stride) p[i] = z;
}

// ---------------- 2. x: f32 -> bf16 (elementwise) ----------------
__global__ void xconvert_kernel(const float4* __restrict__ x,
                                uint2* __restrict__ xb, int n4) {
  int i = blockIdx.x * blockDim.x + threadIdx.x;
  int stride = gridDim.x * blockDim.x;
  for (; i < n4; i += stride) {
    float4 v = x[i];
    uint2 o;
    o.x = (unsigned)f2bf(v.x) | ((unsigned)f2bf(v.y) << 16);
    o.y = (unsigned)f2bf(v.z) | ((unsigned)f2bf(v.w) << 16);
    xb[i] = o;
  }
}

// ---------------- 3. weights: f32 [e][k][n] -> bf16 transposed [e][n][k] ----
// 64x64 tile per block, 256 threads, LDS staging so both sides are coalesced.
__global__ __launch_bounds__(256) void wconvert_kernel(
    const float* __restrict__ w, unsigned short* __restrict__ wt) {
  __shared__ unsigned short t[64][72];  // [n][k], padded row
  const int tid = threadIdx.x;
  const int blk = blockIdx.x;
  const int e  = blk >> 8;         // 256 tiles per expert
  const int tk = (blk >> 4) & 15;  // tile k
  const int tn = blk & 15;         // tile n
  const float* src = w + ((size_t)e * PD + tk * 64) * PO + tn * 64;
  const int r  = tid >> 4;         // 0..15
  const int c4 = (tid & 15) * 4;   // 0..60
#pragma unroll
  for (int it = 0; it < 4; ++it) {
    const int row = r + it * 16;   // k within tile
    float4 v = *(const float4*)(src + (size_t)row * PO + c4);
    t[c4 + 0][row] = f2bf(v.x);
    t[c4 + 1][row] = f2bf(v.y);
    t[c4 + 2][row] = f2bf(v.z);
    t[c4 + 3][row] = f2bf(v.w);
  }
  __syncthreads();
  unsigned short* dst = wt + ((size_t)e * PO + tn * 64) * PD + tk * 64;
  const int rr = tid >> 2;         // n within tile: 0..63
  const int cc = (tid & 3) * 16;   // k chunk: 0,16,32,48
  uint4* q = (uint4*)(dst + (size_t)rr * PD + cc);
  const uint4* s = (const uint4*)&t[rr][cc];
  q[0] = s[0];
  q[1] = s[1];
}

// ---------------- 4. gating logits ----------------
// 128 threads, 8 tokens per block. logits stored transposed: [(b*E+e)*S + s]
__global__ __launch_bounds__(128) void gating_kernel(
    const float* __restrict__ x, const float* __restrict__ Wg,
    const float* __restrict__ bg, float* __restrict__ logits) {
  __shared__ float xs[8 * PD];
  const int tid = threadIdx.x;
  const int s0 = blockIdx.x * 8;  // global token start
  const float* src = x + (size_t)s0 * PD;
  for (int i = tid; i < 8 * PD / 4; i += 128)
    *(float4*)&xs[i * 4] = *(const float4*)&src[i * 4];
  __syncthreads();
  const int e = tid & 15;
  const int tloc = tid >> 4;  // 0..7
  float acc = bg[e];
  const float* xr = &xs[tloc * PD];
#pragma unroll 8
  for (int d = 0; d < PD; ++d) acc += xr[d] * Wg[d * PE + e];
  const int gt = s0 + tloc;
  const int b = gt >> 11;      // /S
  const int s = gt & (PS - 1);
  logits[(size_t)(b * PE + e) * PS + s] = acc;
}

// ---------------- 5. softmax over S + top-K via bitonic sort ----------------
__global__ __launch_bounds__(1024) void softmax_topk_kernel(
    const float* __restrict__ logits, int* __restrict__ idxw,
    float* __restrict__ valsw) {
  __shared__ float v[PS];
  __shared__ int   id[PS];
  __shared__ float red[1024];
  const int tid = threadIdx.x;
  const int row = blockIdx.x;  // (b*E+e)
  const float* src = logits + (size_t)row * PS;
  float a0 = src[tid], a1 = src[tid + 1024];
  id[tid] = tid; id[tid + 1024] = tid + 1024;
  red[tid] = fmaxf(a0, a1);
  __syncthreads();
  for (int off = 512; off > 0; off >>= 1) {
    if (tid < off) red[tid] = fmaxf(red[tid], red[tid + off]);
    __syncthreads();
  }
  const float mx = red[0];
  __syncthreads();
  const float e0 = __expf(a0 - mx), e1 = __expf(a1 - mx);
  red[tid] = e0 + e1;
  __syncthreads();
  for (int off = 512; off > 0; off >>= 1) {
    if (tid < off) red[tid] += red[tid + off];
    __syncthreads();
  }
  const float inv = 1.0f / red[0];
  v[tid] = e0 * inv; v[tid + 1024] = e1 * inv;
  __syncthreads();
  // bitonic sort, final order descending (ties: lower index first)
  for (int k = 2; k <= PS; k <<= 1) {
    for (int j = k >> 1; j > 0; j >>= 1) {
      for (int t = tid; t < PS; t += 1024) {
        int ixj = t ^ j;
        if (ixj > t) {
          float va = v[t], vb = v[ixj];
          int ia = id[t], ib = id[ixj];
          bool desc = ((t & k) == 0);
          bool sw = desc ? (va < vb || (va == vb && ia > ib))
                         : (va > vb || (va == vb && ia < ib));
          if (sw) { v[t] = vb; v[ixj] = va; id[t] = ib; id[ixj] = ia; }
        }
      }
      __syncthreads();
    }
  }
  if (tid < PK) {
    idxw[row * PK + tid]  = id[tid];
    valsw[row * PK + tid] = v[tid];
  }
}

// ---------------- 6-8. expert GEMMs via v_wmma_f32_16x16x32_bf16 ----------
// All operands pre-converted to bf16; staging is pure b128 copies.
// MODE 1: A = gather(xbf, idx),  out = relu(A@w1 + b1) -> h bf16
// MODE 2: A = h1,                out = relu(A@w2 + b2) -> h bf16
// MODE 3: A = h2,                out = (A@w3 + b3) * vals, atomicAdd scatter
// Block: 256 thr (8 waves), tile 128x128, K-step 32, double-buffered LDS.
template <int MODE>
__global__ __launch_bounds__(256) void expert_gemm_kernel(
    const unsigned short* __restrict__ xbf,  // [B][S][D] bf16 (MODE 1)
    const unsigned short* __restrict__ hin,  // [BE][K][O] bf16 (MODE 2/3)
    const unsigned short* __restrict__ wt,   // [E][n][k] bf16 (transposed)
    const float* __restrict__ bias,
    const int* __restrict__ idxw, const float* __restrict__ valsw,
    unsigned short* __restrict__ hout, float* __restrict__ dout) {
  __shared__ unsigned short As[2][128 * 32];  // [m][k] bf16
  __shared__ unsigned short Bs[2][128 * 32];  // [n][k] bf16
  __shared__ int   rowsL[128];
  __shared__ float valsL[128];

  const int tid = threadIdx.x;
  const int blk = blockIdx.x;
  const int be = blk >> 4;
  const int tt = blk & 15;
  const int m0 = (tt >> 3) * 128;
  const int n0 = (tt & 7) * 128;
  const int b = be >> 4;   // /E
  const int e = be & 15;

  if (MODE == 3 && tid < 128) {
    rowsL[tid] = idxw[be * PK + m0 + tid];
    valsL[tid] = valsw[be * PK + m0 + tid];
  }

  // per-thread staging source rows (fixed for whole K loop)
  const int ar = tid >> 1;          // local row 0..127 (A: m, B: n)
  const int ac = (tid & 1) * 16;    // 16-element half-chunk
  const unsigned short* asrc;
  if (MODE == 1) {
    const int grow = idxw[be * PK + m0 + ar];  // gather token index
    asrc = xbf + ((size_t)b * PS + grow) * PD + ac;
  } else {
    asrc = hin + ((size_t)be * PK + m0 + ar) * PO + ac;
  }
  const unsigned short* bsrc =
      wt + ((size_t)e * PO + n0 + ar) * PD + ac;
  unsigned short* adst0 = &As[0][ar * 32 + ac];
  unsigned short* adst1 = &As[1][ar * 32 + ac];
  unsigned short* bdst0 = &Bs[0][ar * 32 + ac];
  unsigned short* bdst1 = &Bs[1][ar * 32 + ac];

  const int lane = tid & 31;
  const int wv = tid >> 5;
  const int wm = wv >> 2, wn = wv & 3;   // 2x4 wave grid
  const int lh = lane & 15, hihalf = lane >> 4;
  const int koff = hihalf * 8;    // A frag: lanes 16-31 hold K +8
  const int kb   = hihalf * 8;    // B frag: lanes 16-31 hold K +8

  v8f acc[4][2];
#pragma unroll
  for (int mf = 0; mf < 4; ++mf)
#pragma unroll
    for (int nf = 0; nf < 2; ++nf) acc[mf][nf] = v8f{};

  // ---- prologue: stage K-step 0 into buffer 0 ----
  {
    const uint4* pa = (const uint4*)asrc;
    const uint4* pb = (const uint4*)bsrc;
    uint4 a0 = pa[0], a1 = pa[1], b0 = pb[0], b1 = pb[1];
    ((uint4*)adst0)[0] = a0; ((uint4*)adst0)[1] = a1;
    ((uint4*)bdst0)[0] = b0; ((uint4*)bdst0)[1] = b1;
  }
  __syncthreads();

  const int NK = PD / 32;  // 32 K-steps
  for (int ks = 0; ks < NK; ++ks) {
    const int buf = ks & 1;
    // issue next tile's global loads (overlap with WMMA below)
    uint4 a0, a1, b0, b1;
    if (ks + 1 < NK) {
      const uint4* pa = (const uint4*)(asrc + (ks + 1) * 32);
      const uint4* pb = (const uint4*)(bsrc + (ks + 1) * 32);
      a0 = pa[0]; a1 = pa[1]; b0 = pb[0]; b1 = pb[1];
    }
    // ---- compute on current buffer ----
    {
      v16bf afrag[4], bfrag[2];
#pragma unroll
      for (int mf = 0; mf < 4; ++mf) {
        const int m = wm * 64 + mf * 16 + lh;
        afrag[mf] = load_frag(&As[buf][m * 32 + koff],
                              &As[buf][m * 32 + koff + 16]);
      }
#pragma unroll
      for (int nf = 0; nf < 2; ++nf) {
        const int n = wn * 32 + nf * 16 + lh;
        bfrag[nf] = load_frag(&Bs[buf][n * 32 + kb],
                              &Bs[buf][n * 32 + kb + 16]);
      }
#pragma unroll
      for (int mf = 0; mf < 4; ++mf)
#pragma unroll
        for (int nf = 0; nf < 2; ++nf)
          acc[mf][nf] = __builtin_amdgcn_wmma_f32_16x16x32_bf16(
              false, afrag[mf], false, bfrag[nf], (short)0, acc[mf][nf],
              false, false);
    }
    // store staged registers into the other buffer
    if (ks + 1 < NK) {
      unsigned short* ad = buf ? adst0 : adst1;
      unsigned short* bd = buf ? bdst0 : bdst1;
      ((uint4*)ad)[0] = a0; ((uint4*)ad)[1] = a1;
      ((uint4*)bd)[0] = b0; ((uint4*)bd)[1] = b1;
    }
    __syncthreads();
  }

  // ---- epilogue ----
  const float bscal = bias[e];  // b* is (E,1): scalar per expert
#pragma unroll
  for (int mf = 0; mf < 4; ++mf) {
#pragma unroll
    for (int nf = 0; nf < 2; ++nf) {
#pragma unroll
      for (int r = 0; r < 8; ++r) {
        const int m = wm * 64 + mf * 16 + r + hihalf * 8;  // C/D layout
        const int n = n0 + wn * 32 + nf * 16 + lh;
        float vvv = acc[mf][nf][r] + bscal;
        if (MODE != 3) {
          vvv = vvv > 0.f ? vvv : 0.f;
          hout[((size_t)be * PK + m0 + m) * PO + n] = f2bf(vvv);
        } else {
          atomicAdd(&dout[((size_t)b * PS + rowsL[m]) * PO + n],
                    vvv * valsL[m]);
        }
      }
    }
  }
}

// ---------------- host launcher ----------------
extern "C" void kernel_launch(void* const* d_in, const int* in_sizes, int n_in,
                              void* d_out, int out_size, void* d_ws,
                              size_t ws_size, hipStream_t stream) {
  const float* x  = (const float*)d_in[0];
  const float* Wg = (const float*)d_in[1];
  const float* bg = (const float*)d_in[2];
  const float* w1 = (const float*)d_in[3];
  const float* b1 = (const float*)d_in[4];
  const float* w2 = (const float*)d_in[5];
  const float* b2 = (const float*)d_in[6];
  const float* w3 = (const float*)d_in[7];
  const float* b3 = (const float*)d_in[8];
  float* out = (float*)d_out;

  char* ws = (char*)d_ws;
  size_t off = 0;
  float* logits = (float*)(ws + off);  off += (size_t)PBE * PS * sizeof(float);
  int*   idxw   = (int*)(ws + off);    off += (size_t)PBE * PK * sizeof(int);
  float* valsw  = (float*)(ws + off);  off += (size_t)PBE * PK * sizeof(float);
  unsigned short* xbf = (unsigned short*)(ws + off);
  off += (size_t)PB * PS * PD * 2;                          // 33.5 MB
  unsigned short* wT1 = (unsigned short*)(ws + off);
  off += (size_t)PE * PD * PO * 2;                          // 33.5 MB
  unsigned short* wT2 = (unsigned short*)(ws + off);
  off += (size_t)PE * PD * PO * 2;
  unsigned short* wT3 = (unsigned short*)(ws + off);
  off += (size_t)PE * PD * PO * 2;
  unsigned short* h1 = (unsigned short*)(ws + off);
  off += (size_t)PBE * PK * PO * 2;                         // 64 MB
  unsigned short* h2 = (unsigned short*)(ws + off);         // 64 MB

  // 1. zero output (harness poisons d_out)
  zero_kernel<<<2048, 256, 0, stream>>>((float4*)out, PB * PS * PO / 4);
  // 2. precision prep: x -> bf16; weights -> bf16 transposed [e][n][k]
  xconvert_kernel<<<2048, 256, 0, stream>>>((const float4*)x, (uint2*)xbf,
                                            PB * PS * PD / 4);
  wconvert_kernel<<<PE * 256, 256, 0, stream>>>(w1, wT1);
  wconvert_kernel<<<PE * 256, 256, 0, stream>>>(w2, wT2);
  wconvert_kernel<<<PE * 256, 256, 0, stream>>>(w3, wT3);
  // 3. gating logits
  gating_kernel<<<PB * PS / 8, 128, 0, stream>>>(x, Wg, bg, logits);
  // 4. softmax over S + top-256 per (b,e)
  softmax_topk_kernel<<<PBE, 1024, 0, stream>>>(logits, idxw, valsw);
  // 5-7. expert MLP: 3 chained bf16-WMMA GEMMs
  const int gblocks = PBE * (PK / 128) * (PO / 128);  // 2048
  expert_gemm_kernel<1><<<gblocks, 256, 0, stream>>>(xbf, h1, wT1, b1, idxw,
                                                     valsw, h1, out);
  expert_gemm_kernel<2><<<gblocks, 256, 0, stream>>>(xbf, h1, wT2, b2, idxw,
                                                     valsw, h2, out);
  expert_gemm_kernel<3><<<gblocks, 256, 0, stream>>>(xbf, h2, wT3, b3, idxw,
                                                     valsw, h2, out);
}
